// MMD_loss_3917010174518
// MI455X (gfx1250) — matrix operand order repaired and checked by reading, not compiled, not run on hardware
//
#include <hip/hip_runtime.h>

typedef float  v2f  __attribute__((ext_vector_type(2)));
typedef float  v8f  __attribute__((ext_vector_type(8)));
typedef __bf16 bf16_t;
typedef __bf16 v8bf  __attribute__((ext_vector_type(8)));
typedef __bf16 v16bf __attribute__((ext_vector_type(16)));

#define N_TOT 8192
#define DIM   256
#define BHALF 4096

// ws layout: first 128KB = float scratch, then bf16 hi/lo planes (4MB each)
#define WS_SQ    0        // 8192 row square-norms
#define WS_COLP  8192     // 32*256 column-sum partials
#define WS_COEF  16384    // 5 coefficients (-gamma_i * log2(e))
#define WS_PART  16400    // 2048 per-block partial sums
#define N_PART   2048
#define WS_BF_OFF   (128 * 1024)                       // bytes
#define N_ELEM      ((size_t)N_TOT * DIM)              // 2M elements
#define WS_NEED     (WS_BF_OFF + 2 * N_ELEM * sizeof(bf16_t) * 2)

// ---------------------------------------------------------------- row ||x||^2
__global__ __launch_bounds__(256) void k_rowsq(const float* __restrict__ xs,
                                               const float* __restrict__ xt,
                                               float* __restrict__ sq) {
    int w = threadIdx.x >> 5, lane = threadIdx.x & 31;
    int r = blockIdx.x * 8 + w;
    const float* p = (r < BHALF) ? xs + (size_t)r * DIM
                                 : xt + (size_t)(r - BHALF) * DIM;
    const float4* p4 = (const float4*)(p + lane * 8);
    float4 a = p4[0], b = p4[1];
    float s = a.x*a.x + a.y*a.y + a.z*a.z + a.w*a.w
            + b.x*b.x + b.y*b.y + b.z*b.z + b.w*b.w;
    for (int off = 16; off > 0; off >>= 1) s += __shfl_down(s, off, 32);
    if (lane == 0) sq[r] = s;
}

// ------------------------------------------------- column-sum partials (32 blocks)
__global__ __launch_bounds__(256) void k_colpart(const float* __restrict__ xs,
                                                 const float* __restrict__ xt,
                                                 float* __restrict__ colp) {
    int t = threadIdx.x;
    int rb = blockIdx.x * 256;
    const float* p = (rb < BHALF) ? xs + (size_t)rb * DIM
                                  : xt + (size_t)(rb - BHALF) * DIM;
    float acc = 0.0f;
    for (int r = 0; r < 256; ++r) acc += p[(size_t)r * DIM + t];
    colp[blockIdx.x * 256 + t] = acc;
}

// ----------------------------------- bandwidth -> 5 exp2 coefficients (1 block)
__global__ __launch_bounds__(256) void k_bw(const float* __restrict__ sq,
                                            const float* __restrict__ colp,
                                            float* __restrict__ coef) {
    __shared__ float s1[256], s2[256];
    int t = threadIdx.x;
    float cs = 0.0f;
    for (int b = 0; b < 32; ++b) cs += colp[b * 256 + t];
    float ssq = 0.0f;
    for (int i = 0; i < 32; ++i) ssq += sq[t + 256 * i];
    s1[t] = cs * cs;
    s2[t] = ssq;
    __syncthreads();
    for (int off = 128; off > 0; off >>= 1) {
        if (t < off) { s1[t] += s1[t + off]; s2[t] += s2[t + off]; }
        __syncthreads();
    }
    if (t == 0) {
        float sumL2 = 2.0f * (float)N_TOT * s2[0] - 2.0f * s1[0];
        float denom = (float)((long long)N_TOT * N_TOT - N_TOT);
        float bw = (sumL2 / denom) * 0.25f;   // / KERNEL_MUL^(KERNEL_NUM//2) = 4
        float g = 1.0f / bw;
        const float LOG2E = 1.4426950408889634f;
        #pragma unroll
        for (int i = 0; i < 5; ++i) { coef[i] = -g * LOG2E; g *= 0.5f; }
    }
}

// ------------------------------------- split f32 -> bf16 hi + bf16 lo planes
__global__ __launch_bounds__(256) void k_cvt(const float* __restrict__ xs,
                                             const float* __restrict__ xt,
                                             bf16_t* __restrict__ hi,
                                             bf16_t* __restrict__ lo) {
    size_t i8 = ((size_t)blockIdx.x * 256 + threadIdx.x) * 8;
    const float* src = (i8 < (size_t)BHALF * DIM) ? xs + i8
                                                  : xt + (i8 - (size_t)BHALF * DIM);
    float4 a = *(const float4*)(src);
    float4 b = *(const float4*)(src + 4);
    float v[8] = {a.x, a.y, a.z, a.w, b.x, b.y, b.z, b.w};
    v8bf hv, lv;
    #pragma unroll
    for (int j = 0; j < 8; ++j) {
        bf16_t hb = (bf16_t)v[j];
        float  r  = v[j] - (float)hb;
        hv[j] = hb;
        lv[j] = (bf16_t)r;
    }
    *(v8bf*)(hi + i8) = hv;
    *(v8bf*)(lo + i8) = lv;
}

// ---------------- bf16x3 fused Gram/WMMA + RBF kernels + signed reduction
// G ~= Ahi*Bhi + Ahi*Blo + Alo*Bhi  (lo*lo term ~2^-16 relative, dropped)
__global__ __launch_bounds__(256) void k_mmd_bf3(const bf16_t* __restrict__ hi,
                                                 const bf16_t* __restrict__ lo,
                                                 const float* __restrict__ sq,
                                                 const float* __restrict__ coef,
                                                 float* __restrict__ part) {
    __shared__ float sred[8];
    int lane = threadIdx.x & 31;
    int w    = threadIdx.x >> 5;
    int r0 = blockIdx.y * 256 + (w >> 1) * 64;
    int c0 = blockIdx.x * 128 + (w & 1)  * 64;

    float wsum = 0.0f;
    if (r0 <= c0) {                       // wave-uniform: EXEC all-ones inside
        int m = lane & 15;                // M for A rows / N for B cols
        int h = lane >> 4;                // lane-half selects K sub-range
        v8f acc[4][4] = {};

        #pragma unroll 2
        for (int k = 0; k < DIM; k += 32) {
            // B fragments: per lane one contiguous 16-K run at K = k + 16*h
            v16bf bh[4], bl[4];
            #pragma unroll
            for (int tj = 0; tj < 4; ++tj) {
                size_t o = (size_t)(c0 + tj * 16 + m) * DIM + k + 16 * h;
                bh[tj] = *(const v16bf*)(hi + o);
                bl[tj] = *(const v16bf*)(lo + o);
            }
            #pragma unroll
            for (int ti = 0; ti < 4; ++ti) {
                // A fragments: two contiguous 8-K octets at K = k+16h, k+16h+8
                size_t ro = (size_t)(r0 + ti * 16 + m) * DIM + k + 16 * h;
                v8bf a0 = *(const v8bf*)(hi + ro);
                v8bf a1 = *(const v8bf*)(hi + ro + 8);
                v8bf l0 = *(const v8bf*)(lo + ro);
                v8bf l1 = *(const v8bf*)(lo + ro + 8);
                v16bf aH = __builtin_shufflevector(a0, a1, 0,1,2,3,4,5,6,7,
                                                   8,9,10,11,12,13,14,15);
                v16bf aL = __builtin_shufflevector(l0, l1, 0,1,2,3,4,5,6,7,
                                                   8,9,10,11,12,13,14,15);
                #pragma unroll
                for (int tj = 0; tj < 4; ++tj) {
                    acc[ti][tj] = __builtin_amdgcn_wmma_f32_16x16x32_bf16(
                        false, aH, false, bh[tj], (short)0, acc[ti][tj], false, false);
                    acc[ti][tj] = __builtin_amdgcn_wmma_f32_16x16x32_bf16(
                        false, aH, false, bl[tj], (short)0, acc[ti][tj], false, false);
                    acc[ti][tj] = __builtin_amdgcn_wmma_f32_16x16x32_bf16(
                        false, aL, false, bh[tj], (short)0, acc[ti][tj], false, false);
                }
            }
        }

        float nc0 = coef[0], nc1 = coef[1], nc2 = coef[2],
              nc3 = coef[3], nc4 = coef[4];

        #pragma unroll
        for (int ti = 0; ti < 4; ++ti) {
            int rowBase = r0 + ti * 16 + (h << 3);
            float4 sr0 = *(const float4*)(sq + rowBase);
            float4 sr1 = *(const float4*)(sq + rowBase + 4);
            float srow[8] = {sr0.x, sr0.y, sr0.z, sr0.w,
                             sr1.x, sr1.y, sr1.z, sr1.w};
            bool rs = rowBase < BHALF;
            #pragma unroll
            for (int tj = 0; tj < 4; ++tj) {
                int col = c0 + tj * 16 + m;
                float scol = sq[col];
                bool csn = (c0 + tj * 16) < BHALF;
                float sgn = (rs == csn) ? 1.0f : -1.0f;
                v8f g = acc[ti][tj];
                #pragma unroll
                for (int v = 0; v < 8; ++v) {
                    int row = rowBase + v;
                    float wgt = (r0 < c0) ? 2.0f
                              : ((row < col) ? 2.0f : ((row == col) ? 1.0f : 0.0f));
                    float l2 = fmaxf(srow[v] + scol - 2.0f * g[v], 0.0f);
                    float kern = exp2f(l2 * nc0) + exp2f(l2 * nc1) + exp2f(l2 * nc2)
                               + exp2f(l2 * nc3) + exp2f(l2 * nc4);
                    wsum += sgn * wgt * kern;
                }
            }
        }
    }

    for (int off = 16; off > 0; off >>= 1) wsum += __shfl_down(wsum, off, 32);
    if (lane == 0) sred[w] = wsum;
    __syncthreads();
    if (threadIdx.x == 0) {
        float t = 0.0f;
        for (int i = 0; i < 8; ++i) t += sred[i];
        part[blockIdx.y * gridDim.x + blockIdx.x] = t;
    }
}

// ------------------- f32-WMMA fallback (used when ws too small for bf16 planes)
__global__ __launch_bounds__(256) void k_mmd(const float* __restrict__ xs,
                                             const float* __restrict__ xt,
                                             const float* __restrict__ sq,
                                             const float* __restrict__ coef,
                                             float* __restrict__ part) {
    __shared__ float sred[8];
    int lane = threadIdx.x & 31;
    int w    = threadIdx.x >> 5;
    int r0 = blockIdx.y * 256 + (w >> 1) * 64;
    int c0 = blockIdx.x * 128 + (w & 1)  * 64;

    float wsum = 0.0f;
    if (r0 <= c0) {
        const float* aBase = (r0 < BHALF) ? xs + (size_t)r0 * DIM
                                          : xt + (size_t)(r0 - BHALF) * DIM;
        const float* bBase = (c0 < BHALF) ? xs + (size_t)c0 * DIM
                                          : xt + (size_t)(c0 - BHALF) * DIM;
        int m    = lane & 15;
        int kofs = (lane >> 4) << 1;
        v8f acc[4][4] = {};

        #pragma unroll 2
        for (int k = 0; k < DIM; k += 4) {
            v2f af[4], bf[4];
            #pragma unroll
            for (int ti = 0; ti < 4; ++ti)
                af[ti] = *(const v2f*)(aBase + (size_t)(ti * 16 + m) * DIM + k + kofs);
            #pragma unroll
            for (int tj = 0; tj < 4; ++tj)
                bf[tj] = *(const v2f*)(bBase + (size_t)(tj * 16 + m) * DIM + k + kofs);
            #pragma unroll
            for (int ti = 0; ti < 4; ++ti) {
                #pragma unroll
                for (int tj = 0; tj < 4; ++tj) {
                    acc[ti][tj] = __builtin_amdgcn_wmma_f32_16x16x4_f32(
                        false, af[ti], false, bf[tj],
                        (short)0, acc[ti][tj], false, false);
                }
            }
        }

        float nc0 = coef[0], nc1 = coef[1], nc2 = coef[2],
              nc3 = coef[3], nc4 = coef[4];

        #pragma unroll
        for (int ti = 0; ti < 4; ++ti) {
            int rowBase = r0 + ti * 16 + ((lane >> 4) << 3);
            float4 sr0 = *(const float4*)(sq + rowBase);
            float4 sr1 = *(const float4*)(sq + rowBase + 4);
            float srow[8] = {sr0.x, sr0.y, sr0.z, sr0.w,
                             sr1.x, sr1.y, sr1.z, sr1.w};
            bool rs = rowBase < BHALF;
            #pragma unroll
            for (int tj = 0; tj < 4; ++tj) {
                int col = c0 + tj * 16 + (lane & 15);
                float scol = sq[col];
                bool csn = (c0 + tj * 16) < BHALF;
                float sgn = (rs == csn) ? 1.0f : -1.0f;
                v8f g = acc[ti][tj];
                #pragma unroll
                for (int v = 0; v < 8; ++v) {
                    int row = rowBase + v;
                    float wgt = (r0 < c0) ? 2.0f
                              : ((row < col) ? 2.0f : ((row == col) ? 1.0f : 0.0f));
                    float l2 = fmaxf(srow[v] + scol - 2.0f * g[v], 0.0f);
                    float kern = exp2f(l2 * nc0) + exp2f(l2 * nc1) + exp2f(l2 * nc2)
                               + exp2f(l2 * nc3) + exp2f(l2 * nc4);
                    wsum += sgn * wgt * kern;
                }
            }
        }
    }

    for (int off = 16; off > 0; off >>= 1) wsum += __shfl_down(wsum, off, 32);
    if (lane == 0) sred[w] = wsum;
    __syncthreads();
    if (threadIdx.x == 0) {
        float t = 0.0f;
        for (int i = 0; i < 8; ++i) t += sred[i];
        part[blockIdx.y * gridDim.x + blockIdx.x] = t;
    }
}

// ---------------------------------------------------------------- finalize
__global__ __launch_bounds__(256) void k_final(const float* __restrict__ part,
                                               float* __restrict__ out) {
    __shared__ float s[256];
    int t = threadIdx.x;
    float a = 0.0f;
    for (int i = 0; i < N_PART; i += 256) a += part[i + t];
    s[t] = a;
    __syncthreads();
    for (int off = 128; off > 0; off >>= 1) {
        if (t < off) s[t] += s[t + off];
        __syncthreads();
    }
    if (t == 0) out[0] = s[0] / ((float)BHALF * (float)BHALF);
}

extern "C" void kernel_launch(void* const* d_in, const int* in_sizes, int n_in,
                              void* d_out, int out_size, void* d_ws, size_t ws_size,
                              hipStream_t stream) {
    const float* xs = (const float*)d_in[0];
    const float* xt = (const float*)d_in[1];
    float* out  = (float*)d_out;
    float* wsf  = (float*)d_ws;
    float* sq   = wsf + WS_SQ;
    float* colp = wsf + WS_COLP;
    float* coef = wsf + WS_COEF;
    float* part = wsf + WS_PART;

    k_rowsq  <<<N_TOT / 8, 256, 0, stream>>>(xs, xt, sq);
    k_colpart<<<32,        256, 0, stream>>>(xs, xt, colp);
    k_bw     <<<1,         256, 0, stream>>>(sq, colp, coef);

    dim3 grid(N_TOT / 128, N_TOT / 256);   // 64 x 32 blocks

    if (ws_size >= WS_NEED) {
        bf16_t* hi = (bf16_t*)((char*)d_ws + WS_BF_OFF);
        bf16_t* lo = hi + N_ELEM;
        k_cvt    <<<(unsigned)(N_ELEM / 8 / 256), 256, 0, stream>>>(xs, xt, hi, lo);
        k_mmd_bf3<<<grid, 256, 0, stream>>>(hi, lo, sq, coef, part);
    } else {
        k_mmd    <<<grid, 256, 0, stream>>>(xs, xt, sq, coef, part);
    }

    k_final  <<<1, 256, 0, stream>>>(part, out);
}
